// LinearAttention_72060961292739
// MI455X (gfx1250) — matrix operand order repaired and checked by reading, not compiled, not run on hardware
//
#include <hip/hip_runtime.h>

// ---- problem constants (from reference) ----
#define NB      32
#define CDIM    256          // DIM
#define NHEADS  4
#define DHEAD   32
#define HIDDEN  128          // NHEADS*DHEAD
#define NPIX    4096         // 64*64
#define QKVROWS 384          // 3*HIDDEN
#define SCALE_Q 0.1767766952966369f   // 32^-0.5
#define LN_EPS  1e-5f

// Optional CDNA5 async global->LDS path (compile-time probed)
#if defined(__has_builtin)
#if __has_builtin(__builtin_amdgcn_global_load_async_to_lds_b128) && \
    __has_builtin(__builtin_amdgcn_s_wait_asynccnt)
#define HAVE_ASYNC_LDS 1
#endif
#endif
#ifndef HAVE_ASYNC_LDS
#define HAVE_ASYNC_LDS 0
#endif

// pointer types for the async builtin (per hipcc diagnostic: param0 is
// a non-const AS(1) pointer to a 16-byte int vector)
typedef int v4i __attribute__((vector_size(16)));
#define AS_GLOBAL __attribute__((address_space(1)))
#define AS_SHARED __attribute__((address_space(3)))

// ---- CDNA5 WMMA types ----
typedef __attribute__((ext_vector_type(16))) __bf16 v16bf;
typedef __attribute__((ext_vector_type(8)))  float  v8f;

union Frag16 {
  v16bf v;
  unsigned int u[8];
  unsigned short s[16];
};

// f32 -> bf16 round-to-nearest-even
__device__ __forceinline__ unsigned short f2bf(float f) {
  union { float f; unsigned int u; } c; c.f = f;
  unsigned int r = c.u + 0x7FFFu + ((c.u >> 16) & 1u);
  return (unsigned short)(r >> 16);
}

// bf16 16x32 A-matrix K index for VGPR v (u32 pair base), half-wave group g.
// Contiguous K order (matches B): lanes 0-15 -> K=0..15, lanes 16-31 -> K=16..31.
__device__ __forceinline__ int aKbase(int v, int g) {
  return 16 * g + 2 * v;
}

// ============================================================================
// K1: qkv = w_qkv(384x256) x x[b](256x4096), f32 out.
// Block = 256 thr = 8 waves (4 M-tiles x 2 N-groups). Block tile 64x128.
// Each wave: 16x64 -> one A frag reused across 4 WMMAs.
// ============================================================================
__global__ __launch_bounds__(256)
void k_qkv_gemm(const float* __restrict__ x, const float* __restrict__ wqkv,
                float* __restrict__ qkv) {
  __shared__ __align__(16) unsigned short ldsA[64 * 34];    // [m][k]
  __shared__ __align__(16) unsigned short ldsB[128 * 34];   // [n][k] transposed

  const int t = threadIdx.x;
  const int lane = t & 31, w = t >> 5;
  const int mtw = w >> 1, ng = w & 1;
  const int l15 = lane & 15, g = lane >> 4;
  const int n0 = blockIdx.x * 128;
  const int m0 = blockIdx.y * 64;
  const int b  = blockIdx.z;
  const float* xb = x + (size_t)b * CDIM * NPIX;

  v8f acc[4] = {};
  for (int kb = 0; kb < CDIM; kb += 32) {
    // stage A: 64x32 of w_qkv, f32 -> bf16
    #pragma unroll
    for (int i = 0; i < 8; ++i) {
      int j = i * 256 + t;
      int m = j >> 5, k = j & 31;
      ldsA[m * 34 + k] = f2bf(wqkv[(m0 + m) * CDIM + kb + k]);
    }
    // stage B: 32(K) x 128(N) of x, transposed into [n][k]
    #pragma unroll
    for (int i = 0; i < 16; ++i) {
      int j = i * 256 + t;
      int n = j & 127, k = j >> 7;
      ldsB[n * 34 + k] = f2bf(xb[(size_t)(kb + k) * NPIX + n0 + n]);
    }
    __syncthreads();

    Frag16 a;
    const int M = mtw * 16 + l15;
    #pragma unroll
    for (int v = 0; v < 8; ++v)
      a.u[v] = *(const unsigned int*)&ldsA[M * 34 + aKbase(v, g)];
    #pragma unroll
    for (int s2 = 0; s2 < 4; ++s2) {
      Frag16 bf;
      int N = ng * 64 + s2 * 16 + l15;
      #pragma unroll
      for (int j = 0; j < 8; ++j)
        bf.u[j] = *(const unsigned int*)&ldsB[N * 34 + 16 * g + 2 * j];
      acc[s2] = __builtin_amdgcn_wmma_f32_16x16x32_bf16(false, a.v, false, bf.v,
                                                        (short)0, acc[s2], false, false);
    }
    __syncthreads();
  }

  float* outp = qkv + (size_t)b * QKVROWS * NPIX;
  #pragma unroll
  for (int s2 = 0; s2 < 4; ++s2) {
    int N = n0 + ng * 64 + s2 * 16 + l15;
    #pragma unroll
    for (int r = 0; r < 8; ++r) {
      int m = m0 + mtw * 16 + r + 8 * g;   // C/D layout: M = vgpr + 8*halfwave
      outp[(size_t)m * NPIX + N] = acc[s2][r];
    }
  }
}

// ============================================================================
// K2: q softmax over d (32) per column, *SCALE, write bf16 q
// ============================================================================
__global__ __launch_bounds__(256)
void k_softmax_q(const float* __restrict__ qkv, unsigned short* __restrict__ qb) {
  int idx = blockIdx.x * 256 + threadIdx.x;        // over B*NHEADS*NPIX = 524288
  int b   = idx >> 14;                             // NHEADS*NPIX = 16384
  int rem = idx & 16383;
  int h   = rem >> 12;
  int n   = rem & 4095;
  const float* src = qkv + (size_t)b * QKVROWS * NPIX + (size_t)(h * 32) * NPIX + n;
  float v[32];
  float mx = -1e30f;
  #pragma unroll
  for (int d = 0; d < 32; ++d) { v[d] = src[(size_t)d * NPIX]; mx = fmaxf(mx, v[d]); }
  float s = 0.f;
  #pragma unroll
  for (int d = 0; d < 32; ++d) { v[d] = __expf(v[d] - mx); s += v[d]; }
  float inv = SCALE_Q / s;
  unsigned short* dst = qb + (size_t)(b * HIDDEN + h * 32) * NPIX + n;
  #pragma unroll
  for (int d = 0; d < 32; ++d) dst[(size_t)d * NPIX] = f2bf(v[d] * inv);
}

// ============================================================================
// K3: k softmax over n (4096) per row, write bf16 k. One block per row.
// ============================================================================
__global__ __launch_bounds__(256)
void k_softmax_k(const float* __restrict__ qkv, unsigned short* __restrict__ kb) {
  __shared__ float red[256];
  int r = blockIdx.x;                 // b*128 + o
  int b = r >> 7, o = r & 127;
  const float* src = qkv + ((size_t)b * QKVROWS + HIDDEN + o) * NPIX;
  unsigned short* dst = kb + ((size_t)b * HIDDEN + o) * NPIX;
  int t = threadIdx.x;

  float mx = -1e30f;
  for (int n = t; n < NPIX; n += 256) mx = fmaxf(mx, src[n]);
  red[t] = mx; __syncthreads();
  for (int s = 128; s > 0; s >>= 1) { if (t < s) red[t] = fmaxf(red[t], red[t + s]); __syncthreads(); }
  mx = red[0]; __syncthreads();

  float sum = 0.f;
  for (int n = t; n < NPIX; n += 256) sum += __expf(src[n] - mx);
  red[t] = sum; __syncthreads();
  for (int s = 128; s > 0; s >>= 1) { if (t < s) red[t] += red[t + s]; __syncthreads(); }
  float inv = 1.f / red[0];

  for (int n = t; n < NPIX; n += 256) dst[n] = f2bf(__expf(src[n] - mx) * inv);
}

// ============================================================================
// K4: v cast f32 -> bf16 (16,777,216 elements)
// ============================================================================
__global__ __launch_bounds__(256)
void k_cast_v(const float* __restrict__ qkv, unsigned short* __restrict__ vb) {
  int i = blockIdx.x * 256 + threadIdx.x;
  int b = i >> 19;                        // HIDDEN*NPIX = 524288
  int rem = i & ((1 << 19) - 1);
  vb[i] = f2bf(qkv[(size_t)b * (QKVROWS * NPIX) + (size_t)(2 * HIDDEN) * NPIX + rem]);
}

// ============================================================================
// K5: context[bh] (32x32) = k(32x4096) x v^T, K loop over n.
// Block = 128 thr = 4 waves (2x2 tiles). Tiles staged in LDS (each byte read
// from HBM exactly once); staging uses GLOBAL_LOAD_ASYNC_TO_LDS_B128 +
// s_wait_asynccnt when the toolchain exposes it, else plain 16B copies.
// LDS row stride = 40 halves (80 B) keeps every 16B chunk 16B-aligned.
// ============================================================================
__global__ __launch_bounds__(128)
void k_context(const unsigned short* __restrict__ kbuf,
               const unsigned short* __restrict__ vbuf,
               float* __restrict__ ctx) {
  __shared__ __align__(16) unsigned short ldsK[32 * 40];
  __shared__ __align__(16) unsigned short ldsV[32 * 40];
  int bh = blockIdx.x;
  int t = threadIdx.x, lane = t & 31, w = t >> 5;
  int mt = w >> 1, nt = w & 1;
  int l15 = lane & 15, g = lane >> 4;
  const unsigned short* kp = kbuf + (size_t)bh * 32 * NPIX;
  const unsigned short* vp = vbuf + (size_t)bh * 32 * NPIX;
  const int sr = t >> 2, sc = t & 3;     // staging: row 0..31, 16B chunk 0..3
  int d = mt * 16 + l15;                 // A row
  int e = nt * 16 + l15;                 // B column (row of v)

  v8f acc = {};
  for (int kb0 = 0; kb0 < NPIX; kb0 += 32) {
    const unsigned short* gk = kp + (size_t)sr * NPIX + kb0 + sc * 8;
    const unsigned short* gv = vp + (size_t)sr * NPIX + kb0 + sc * 8;
    unsigned short* lk = &ldsK[sr * 40 + sc * 8];
    unsigned short* lv = &ldsV[sr * 40 + sc * 8];
#if HAVE_ASYNC_LDS
    __builtin_amdgcn_global_load_async_to_lds_b128(
        (AS_GLOBAL v4i*)gk, (AS_SHARED v4i*)lk, 0, 0);
    __builtin_amdgcn_global_load_async_to_lds_b128(
        (AS_GLOBAL v4i*)gv, (AS_SHARED v4i*)lv, 0, 0);
    __builtin_amdgcn_s_wait_asynccnt(0);
#else
    *(uint4*)lk = *(const uint4*)gk;
    *(uint4*)lv = *(const uint4*)gv;
#endif
    __syncthreads();
    Frag16 a, bf;
    #pragma unroll
    for (int v = 0; v < 8; ++v)
      a.u[v] = *(const unsigned int*)&ldsK[d * 40 + aKbase(v, g)];
    #pragma unroll
    for (int j = 0; j < 8; ++j)
      bf.u[j] = *(const unsigned int*)&ldsV[e * 40 + 16 * g + 2 * j];
    acc = __builtin_amdgcn_wmma_f32_16x16x32_bf16(false, a.v, false, bf.v,
                                                  (short)0, acc, false, false);
    __syncthreads();
  }
  float* cp = ctx + (size_t)bh * 1024;   // [d][e]
  #pragma unroll
  for (int r = 0; r < 8; ++r)
    cp[(mt * 16 + r + 8 * g) * 32 + e] = acc[r];
}

// ============================================================================
// K6: attn[bh] (32x4096) = ctx^T (32x32) x q (32x4096). Single K-step of 32.
// Block = 128 thr = 4 waves (2 e-tiles x 2 n-tiles); bf16 output.
// ============================================================================
__global__ __launch_bounds__(128)
void k_attn_out(const float* __restrict__ ctx, const unsigned short* __restrict__ qb,
                unsigned short* __restrict__ attn) {
  int bh = blockIdx.y;
  int n0 = blockIdx.x * 32;
  int t = threadIdx.x, lane = t & 31, w = t >> 5;
  int mt = w >> 1, nt = w & 1;
  int l15 = lane & 15, g = lane >> 4;
  const float* cp = ctx + (size_t)bh * 1024;
  const unsigned short* qp = qb + (size_t)bh * 32 * NPIX;
  int e = mt * 16 + l15;                 // A row (output channel within head)
  int n = n0 + nt * 16 + l15;            // B column

  Frag16 a, bf;
  #pragma unroll
  for (int i = 0; i < 16; ++i) {         // A[e][d] = ctx[d][e]
    int v = i >> 1;
    int K = aKbase(v, g) + (i & 1);      // K = d
    a.s[i] = f2bf(cp[K * 32 + e]);
  }
  #pragma unroll
  for (int i = 0; i < 16; ++i)           // B[d][n] = q[d][n]
    bf.s[i] = qp[(size_t)(16 * g + i) * NPIX + n];

  v8f acc = {};
  acc = __builtin_amdgcn_wmma_f32_16x16x32_bf16(false, a.v, false, bf.v,
                                                (short)0, acc, false, false);
  unsigned short* ap = attn + (size_t)bh * 32 * NPIX;
  #pragma unroll
  for (int r = 0; r < 8; ++r)
    ap[(size_t)(mt * 16 + r + 8 * g) * NPIX + n] = f2bf(acc[r]);
}

// ============================================================================
// K7: out = LN( w_out(256x128) x attn[b](128x4096) + b_out ) * gamma + beta.
// Block = 1024 thr = 32 waves (16 M-tiles x 2 N-tiles) -> full 256-channel
// column tile resident in LDS so the channel LayerNorm fuses in-block.
// ============================================================================
__global__ __launch_bounds__(1024)
void k_proj_ln(const unsigned short* __restrict__ attn, const float* __restrict__ wout,
               const float* __restrict__ bout, const float* __restrict__ gamma,
               const float* __restrict__ beta, float* __restrict__ out) {
  __shared__ __align__(16) unsigned short ldsA[256 * 34];
  __shared__ __align__(16) unsigned short ldsB[32 * 34];     // [n][k]
  __shared__ float ldsO[256 * 34];
  __shared__ float mean_s[32], rstd_s[32];

  int t = threadIdx.x, lane = t & 31, w = t >> 5;
  int mt = w >> 1, nt = w & 1;
  int l15 = lane & 15, g = lane >> 4;
  int n0 = blockIdx.x * 32;
  int b  = blockIdx.y;
  const unsigned short* ab = attn + (size_t)b * HIDDEN * NPIX;

  v8f acc = {};
  for (int kb = 0; kb < HIDDEN; kb += 32) {
    #pragma unroll
    for (int i = 0; i < 8; ++i) {              // stage A 256x32
      int j = i * 1024 + t;
      int m = j >> 5, k = j & 31;
      ldsA[m * 34 + k] = f2bf(wout[m * HIDDEN + kb + k]);
    }
    {                                          // stage B 32x32 (bf16 already)
      int n = t & 31, k = t >> 5;
      ldsB[n * 34 + k] = ab[(size_t)(kb + k) * NPIX + n0 + n];
    }
    __syncthreads();
    Frag16 a, bf;
    int M = mt * 16 + l15;
    int N = nt * 16 + l15;
    #pragma unroll
    for (int v = 0; v < 8; ++v)
      a.u[v] = *(const unsigned int*)&ldsA[M * 34 + aKbase(v, g)];
    #pragma unroll
    for (int j = 0; j < 8; ++j)
      bf.u[j] = *(const unsigned int*)&ldsB[N * 34 + 16 * g + 2 * j];
    acc = __builtin_amdgcn_wmma_f32_16x16x32_bf16(false, a.v, false, bf.v,
                                                  (short)0, acc, false, false);
    __syncthreads();
  }

  {                                            // tile -> LDS with bias
    int N = nt * 16 + l15;
    #pragma unroll
    for (int r = 0; r < 8; ++r) {
      int m = mt * 16 + r + 8 * g;
      ldsO[m * 34 + N] = acc[r] + bout[m];
    }
  }
  __syncthreads();

  if (t < 32) {                                // per-column mean/var over 256 ch
    float s = 0.f, s2 = 0.f;
    for (int m = 0; m < 256; ++m) { float xv = ldsO[m * 34 + t]; s += xv; s2 += xv * xv; }
    float mu  = s  * (1.f / 256.f);
    float var = s2 * (1.f / 256.f) - mu * mu;
    mean_s[t] = mu;
    rstd_s[t] = rsqrtf(var + LN_EPS);
  }
  __syncthreads();

  float* ob = out + (size_t)b * CDIM * NPIX;
  #pragma unroll
  for (int i = 0; i < 8; ++i) {
    int j = i * 1024 + t;
    int m = j >> 5, n = j & 31;
    float xv = (ldsO[m * 34 + n] - mean_s[n]) * rstd_s[n] * gamma[m] + beta[m];
    ob[(size_t)m * NPIX + n0 + n] = xv;
  }
}

// ============================================================================
// launcher
// ============================================================================
extern "C" void kernel_launch(void* const* d_in, const int* in_sizes, int n_in,
                              void* d_out, int out_size, void* d_ws, size_t ws_size,
                              hipStream_t stream) {
  (void)in_sizes; (void)n_in; (void)out_size; (void)ws_size;
  const float* x     = (const float*)d_in[0];
  const float* wqkv  = (const float*)d_in[1];
  const float* wout  = (const float*)d_in[2];
  const float* bout  = (const float*)d_in[3];
  const float* gamma = (const float*)d_in[4];
  const float* beta  = (const float*)d_in[5];
  float* out = (float*)d_out;

  // workspace layout (bytes), total ~336 MB
  char* ws = (char*)d_ws;
  float*          qkv  = (float*)ws;                           // 32*384*4096*4 = 201,326,592
  unsigned short* qb   = (unsigned short*)(ws + 201326592);    // 33,554,432
  unsigned short* kbuf = (unsigned short*)(ws + 234881024);    // 33,554,432
  unsigned short* vb   = (unsigned short*)(ws + 268435456);    // 33,554,432
  float*          ctx  = (float*)(ws + 301989888);             // 524,288
  unsigned short* attn = (unsigned short*)(ws + 302514176);    // 33,554,432

  k_qkv_gemm <<<dim3(32, 6, 32),   256,  0, stream>>>(x, wqkv, qkv);
  k_softmax_q<<<dim3(2048),        256,  0, stream>>>(qkv, qb);
  k_softmax_k<<<dim3(4096),        256,  0, stream>>>(qkv, kbuf);
  k_cast_v   <<<dim3(65536),       256,  0, stream>>>(qkv, vb);
  k_context  <<<dim3(128),         128,  0, stream>>>(kbuf, vb, ctx);
  k_attn_out <<<dim3(128, 128),    128,  0, stream>>>(ctx, qb, attn);
  k_proj_ln  <<<dim3(128, 32),     1024, 0, stream>>>(attn, wout, bout, gamma, beta, out);
}
